// AMLModel_14568529068620
// MI455X (gfx1250) — matrix operand order repaired and checked by reading, not compiled, not run on hardware
//
#include <hip/hip_runtime.h>
#include <math.h>

#define E_TOT 500000
#define NN    50000

typedef __attribute__((ext_vector_type(16))) _Float16 v16h;
typedef __attribute__((ext_vector_type(8)))  float    v8f;

union AFrag { v16h v; _Float16 h[16]; unsigned u[8]; };

// ---- WMMA fragment helpers (layouts per CDNA5 ISA 7.12.2, wave32) ----------

// A fragment (16x32 f16) from LDS tile, row-major [16][ldk] halves.
// lane<16: row=lane, K = k0+{0..7, 16..23}; lane>=16: row=lane-16, K = k0+{8..15, 24..31}
__device__ __forceinline__ v16h lds_afrag(const _Float16* lds, int ldk, int k0, int lane) {
    int hl = lane >> 4, m = lane & 15;
    AFrag f;
    const unsigned* row = (const unsigned*)(lds + m * ldk);
#pragma unroll
    for (int v = 0; v < 8; ++v) {
        int kk = k0 + ((v & 4) << 2) + hl * 8 + ((v & 3) << 1); // pairs of K
        f.u[v] = row[kk >> 1];
    }
    return f.v;
}

// B fragment from pre-packed fragment-major f16 weights: one 32B load per lane.
__device__ __forceinline__ v16h pk_bfrag(const _Float16* __restrict__ Wp, int frag,
                                         int lane) {
    return *(const v16h*)(Wp + ((size_t)frag * 32 + lane) * 16);
}

__device__ __forceinline__ v8f wmma_f16(v16h a, v16h b, v8f c) {
    return __builtin_amdgcn_wmma_f32_16x16x32_f16(false, a, false, b, (short)0, c,
                                                  false, false);
}

__device__ __forceinline__ v8f bcast8(float v) {
    v8f c;
#pragma unroll
    for (int r = 0; r < 8; ++r) c[r] = v;
    return c;
}

// ---- weight pre-pack: W f32 [K][ldn] -> fragment-major f16 -----------------
// frag = ks * nnt + nt; per lane 16 contiguous halves (B-matrix 32x16 layout):
// lane&15 = column, lane>=16 holds K = ks*32+{8..15,24..31}, else K = ks*32+{0..7,16..23}.

__global__ __launch_bounds__(32) void pack_w_k(const float* __restrict__ W, int ldn,
                                               int kmax, int nnt,
                                               _Float16* __restrict__ out) {
    int frag = blockIdx.x;
    int ks = frag / nnt, nt = frag % nnt;
    int lane = threadIdx.x;
    int hl = lane >> 4, n = lane & 15;
    int col = nt * 16 + n;
    AFrag f;
#pragma unroll
    for (int v = 0; v < 8; ++v) {
        int k = ks * 32 + ((v & 4) << 2) + hl * 8 + ((v & 3) << 1);
        f.h[2 * v]     = (_Float16)((k     < kmax) ? W[(size_t)k       * ldn + col] : 0.f);
        f.h[2 * v + 1] = (_Float16)((k + 1 < kmax) ? W[(size_t)(k + 1) * ldn + col] : 0.f);
    }
    *(v16h*)(out + ((size_t)frag * 32 + lane) * 16) = f.v;
}

// ---- utility kernels -------------------------------------------------------

__global__ void fset_k(float* __restrict__ p, int n, float v) {
    int i = blockIdx.x * 256 + threadIdx.x;
    if (i < n) p[i] = v;
}

__global__ void deg_k(const int* __restrict__ ei, float* __restrict__ deg) {
    int e = blockIdx.x * 256 + threadIdx.x;
    if (e < E_TOT) atomicAdd(&deg[ei[E_TOT + e]], 1.f);
}

__global__ void tsmax_k(const int* __restrict__ ts, int* __restrict__ out) {
    int e = blockIdx.x * 256 + threadIdx.x;
    if (e < E_TOT) atomicMax(out, ts[e]);
}

// ---- edge encoder: feats[E,38] -> ReLU(.@w1[38,128]) @ w2[128,64] ----------

__global__ __launch_bounds__(32) void edge_encoder_k(
    const float* __restrict__ la,  const float* __restrict__ cr,
    const float* __restrict__ tsp, const float* __restrict__ tg,
    const float* __restrict__ r7,  const float* __restrict__ r30,
    const float* __restrict__ tse, const int* __restrict__ tt,
    const int* __restrict__ bp,    const float* __restrict__ txe,
    const float* __restrict__ bke,
    const _Float16* __restrict__ w1p, const float* __restrict__ b1,
    const _Float16* __restrict__ w2p, const float* __restrict__ b2,
    _Float16* __restrict__ eattr)
{
    __shared__ _Float16 F[16 * 64];    // padded features (38 -> 64)
    __shared__ _Float16 H[16 * 128];   // hidden
    int lane = threadIdx.x;
    int e0 = blockIdx.x * 16;

    for (int i = lane; i < 16 * 64; i += 32) {
        int m = i >> 6, k = i & 63;
        int e = e0 + m;
        float v = 0.f;
        if      (k == 0) v = la[e];
        else if (k == 1) v = cr[e];
        else if (k == 2) v = tsp[e];
        else if (k == 3) v = tg[e];
        else if (k == 4) v = r7[e];
        else if (k == 5) v = r30[e];
        else if (k < 14) v = tse[(size_t)e * 8 + (k - 6)];
        else if (k < 22) v = txe[(size_t)tt[e] * 8 + (k - 14)];
        else if (k < 30) v = bke[(size_t)bp[(size_t)e * 2] * 8 + (k - 22)];
        else if (k < 38) v = bke[(size_t)bp[(size_t)e * 2 + 1] * 8 + (k - 30)];
        F[i] = (_Float16)v;
    }
    __syncthreads();

    int hl = lane >> 4, nn = lane & 15;
    v16h a0 = lds_afrag(F, 64, 0, lane);
    v16h a1 = lds_afrag(F, 64, 32, lane);
    for (int nt = 0; nt < 8; ++nt) {
        v8f c = bcast8(b1[nt * 16 + nn]);
        c = wmma_f16(a0, pk_bfrag(w1p, 0 * 8 + nt, lane), c);
        c = wmma_f16(a1, pk_bfrag(w1p, 1 * 8 + nt, lane), c);
#pragma unroll
        for (int r = 0; r < 8; ++r)
            H[(r + hl * 8) * 128 + nt * 16 + nn] = (_Float16)fmaxf(c[r], 0.f);
    }
    __syncthreads();

    v16h af[4];
#pragma unroll
    for (int ks = 0; ks < 4; ++ks) af[ks] = lds_afrag(H, 128, ks * 32, lane);
    for (int nt = 0; nt < 4; ++nt) {
        v8f c = bcast8(b2[nt * 16 + nn]);
#pragma unroll
        for (int ks = 0; ks < 4; ++ks)
            c = wmma_f16(af[ks], pk_bfrag(w2p, ks * 4 + nt, lane), c);
#pragma unroll
        for (int r = 0; r < 8; ++r)
            eattr[(size_t)(e0 + r + hl * 8) * 64 + nt * 16 + nn] = (_Float16)c[r];
    }
}

// ---- SAGE message: ReLU([h[src](128) || eattr(64)] @ Wm[192,128]) scatter ->
//      atomicAdd into s[dst] -------------------------------------------------

__global__ __launch_bounds__(32) void sage_msg_k(
    const float* __restrict__ h, const _Float16* __restrict__ eattr,
    const int* __restrict__ ei, const _Float16* __restrict__ wmp,
    const float* __restrict__ bm, float* __restrict__ sacc)
{
    __shared__ _Float16 A[16 * 192];
    int lane = threadIdx.x;
    int e0 = blockIdx.x * 16;

    for (int i = lane; i < 16 * 192; i += 32) {
        int m = i / 192, k = i % 192;
        int e = e0 + m;
        _Float16 v;
        if (k < 128) v = (_Float16)h[(size_t)ei[e] * 128 + k];
        else         v = eattr[(size_t)e * 64 + (k - 128)];
        A[i] = v;
    }
    __syncthreads();

    int hl = lane >> 4, nn = lane & 15;
    v16h af[6];
#pragma unroll
    for (int ks = 0; ks < 6; ++ks) af[ks] = lds_afrag(A, 192, ks * 32, lane);
    int dsti[8];
#pragma unroll
    for (int r = 0; r < 8; ++r) dsti[r] = ei[E_TOT + e0 + r + hl * 8];

    for (int nt = 0; nt < 8; ++nt) {
        v8f c = bcast8(bm[nt * 16 + nn]);
#pragma unroll
        for (int ks = 0; ks < 6; ++ks)
            c = wmma_f16(af[ks], pk_bfrag(wmp, ks * 8 + nt, lane), c);
#pragma unroll
        for (int r = 0; r < 8; ++r)
            atomicAdd(&sacc[(size_t)dsti[r] * 128 + nt * 16 + nn], fmaxf(c[r], 0.f));
    }
}

// ---- SAGE update: ReLU([h(128) || s/deg(128)] @ Wu[256,nout]) --------------

__global__ __launch_bounds__(32) void sage_upd_k(
    const float* __restrict__ h, const float* __restrict__ sacc,
    const float* __restrict__ deg, const _Float16* __restrict__ wup,
    const float* __restrict__ bu, int nout, float* __restrict__ out)
{
    __shared__ _Float16 A[16 * 256];
    int lane = threadIdx.x;
    int v0 = blockIdx.x * 16;

    for (int i = lane; i < 16 * 256; i += 32) {
        int m = i >> 8, k = i & 255;
        int node = v0 + m;
        float v;
        if (k < 128) v = h[(size_t)node * 128 + k];
        else         v = sacc[(size_t)node * 128 + (k - 128)] / (deg[node] + 1e-6f);
        A[i] = (_Float16)v;
    }
    __syncthreads();

    int hl = lane >> 4, nn = lane & 15;
    v16h af[8];
#pragma unroll
    for (int ks = 0; ks < 8; ++ks) af[ks] = lds_afrag(A, 256, ks * 32, lane);

    int nnt = nout >> 4;
    for (int nt = 0; nt < nnt; ++nt) {
        v8f c = bcast8(bu[nt * 16 + nn]);
#pragma unroll
        for (int ks = 0; ks < 8; ++ks)
            c = wmma_f16(af[ks], pk_bfrag(wup, ks * nnt + nt, lane), c);
#pragma unroll
        for (int r = 0; r < 8; ++r)
            out[(size_t)(v0 + r + hl * 8) * nout + nt * 16 + nn] = fmaxf(c[r], 0.f);
    }
}

// ---- edge classifier: [emb[src]||emb[dst]||eattr](192) -> 64 -> 1 ----------

__global__ __launch_bounds__(32) void edge_cls_k(
    const float* __restrict__ emb, const _Float16* __restrict__ eattr,
    const int* __restrict__ ei, const _Float16* __restrict__ w1p,
    const float* __restrict__ b1, const float* __restrict__ w2,
    const float* __restrict__ b2, float* __restrict__ logits,
    float* __restrict__ probs)
{
    __shared__ _Float16 A[16 * 192];
    __shared__ float H[16 * 64];
    int lane = threadIdx.x;
    int e0 = blockIdx.x * 16;

    for (int i = lane; i < 16 * 192; i += 32) {
        int m = i / 192, k = i % 192;
        int e = e0 + m;
        float v;
        if      (k < 64)  v = emb[(size_t)ei[e] * 64 + k];
        else if (k < 128) v = emb[(size_t)ei[E_TOT + e] * 64 + (k - 64)];
        else              v = (float)eattr[(size_t)e * 64 + (k - 128)];
        A[i] = (_Float16)v;
    }
    __syncthreads();

    int hl = lane >> 4, nn = lane & 15;
    v16h af[6];
#pragma unroll
    for (int ks = 0; ks < 6; ++ks) af[ks] = lds_afrag(A, 192, ks * 32, lane);
    for (int nt = 0; nt < 4; ++nt) {
        v8f c = bcast8(b1[nt * 16 + nn]);
#pragma unroll
        for (int ks = 0; ks < 6; ++ks)
            c = wmma_f16(af[ks], pk_bfrag(w1p, ks * 4 + nt, lane), c);
#pragma unroll
        for (int r = 0; r < 8; ++r)
            H[(r + hl * 8) * 64 + nt * 16 + nn] = fmaxf(c[r], 0.f);
    }
    __syncthreads();

    if (lane < 16) {
        float acc = b2[0];
        for (int j = 0; j < 64; ++j) acc += H[lane * 64 + j] * w2[j];
        logits[e0 + lane] = acc;
        probs[e0 + lane] = 1.f / (1.f + __expf(-acc));
    }
}

// ---- edge-signal aggregation (atomics) -------------------------------------
// acc layout (11 arrays of NN): 0 sum_p, 1 cnt, 2 maxp, 3 cnt_hi, 4 sum_pd,
// 5 sum_d, 6 cnt_hi30, 7 max30, 8 sum_p30, 9 sum_30, 10 minage(9999 init)

__global__ void eagg_k(const float* __restrict__ probs, const int* __restrict__ ei,
                       const int* __restrict__ uts, const int* __restrict__ nowp,
                       float* __restrict__ acc)
{
    int e = blockIdx.x * 256 + threadIdx.x;
    if (e >= E_TOT) return;
    float p = probs[e];
    float age = fmaxf((float)(*nowp) - (float)uts[e], 0.f);
    float decay = __expf(-age / (30.f * 86400.f));
    bool high = (p >= 0.7f);
    bool last30 = (age <= 30.f * 86400.f);
    float aged = age / 86400.f;
    int pb = __float_as_int(p);
    int u0 = ei[e], u1 = ei[E_TOT + e];
#pragma unroll
    for (int s = 0; s < 2; ++s) {
        int u = s ? u1 : u0;
        atomicAdd(&acc[0 * NN + u], p);
        atomicAdd(&acc[1 * NN + u], 1.f);
        atomicMax((int*)&acc[2 * NN + u], pb);
        atomicAdd(&acc[4 * NN + u], p * decay);
        atomicAdd(&acc[5 * NN + u], decay);
        if (high) {
            atomicAdd(&acc[3 * NN + u], 1.f);
            atomicMin((int*)&acc[10 * NN + u], __float_as_int(aged));
        }
        if (last30) {
            if (high) atomicAdd(&acc[6 * NN + u], 1.f);
            atomicMax((int*)&acc[7 * NN + u], pb);
            atomicAdd(&acc[8 * NN + u], p);
            atomicAdd(&acc[9 * NN + u], 1.f);
        }
    }
}

__global__ void agg_fin_k(const float* __restrict__ acc, float* __restrict__ out) {
    int n = blockIdx.x * 256 + threadIdx.x;
    if (n >= NN) return;
    float cnt = acc[1 * NN + n];
    out[n * 8 + 0] = acc[0 * NN + n] / (cnt + 1e-6f);
    out[n * 8 + 1] = acc[2 * NN + n];
    out[n * 8 + 2] = log1pf(acc[3 * NN + n]);
    out[n * 8 + 3] = acc[4 * NN + n] / (acc[5 * NN + n] + 1e-6f);
    out[n * 8 + 4] = log1pf(acc[6 * NN + n]);
    out[n * 8 + 5] = acc[7 * NN + n];
    out[n * 8 + 6] = acc[8 * NN + n] / (acc[9 * NN + n] + 1e-6f);
    float ma = fminf(acc[10 * NN + n], 9999.f);
    out[n * 8 + 7] = log1pf(fminf(ma, 90.f)) / log1pf(90.f);
}

// ---- node classifier: [emb(64)||agg(8)](72, pad 96) -> 64 (BN,ReLU) -> 1 ---

__global__ __launch_bounds__(32) void node_cls_k(
    const float* __restrict__ emb, const float* __restrict__ agg,
    const _Float16* __restrict__ w1p, const float* __restrict__ b1,
    const float* __restrict__ gamma, const float* __restrict__ beta,
    const float* __restrict__ mean, const float* __restrict__ var,
    const float* __restrict__ w2, const float* __restrict__ b2,
    float* __restrict__ logits)
{
    __shared__ _Float16 A[16 * 96];
    __shared__ float H[16 * 64];
    int lane = threadIdx.x;
    int v0 = blockIdx.x * 16;

    for (int i = lane; i < 16 * 96; i += 32) {
        int m = i / 96, k = i % 96;
        int node = v0 + m;
        float v = 0.f;
        if (k < 64)      v = emb[(size_t)node * 64 + k];
        else if (k < 72) v = agg[(size_t)node * 8 + (k - 64)];
        A[i] = (_Float16)v;
    }
    __syncthreads();

    int hl = lane >> 4, nn = lane & 15;
    v16h af[3];
#pragma unroll
    for (int ks = 0; ks < 3; ++ks) af[ks] = lds_afrag(A, 96, ks * 32, lane);
    for (int nt = 0; nt < 4; ++nt) {
        int col = nt * 16 + nn;
        v8f c = bcast8(b1[col]);
#pragma unroll
        for (int ks = 0; ks < 3; ++ks)
            c = wmma_f16(af[ks], pk_bfrag(w1p, ks * 4 + nt, lane), c);
        float g = gamma[col] * rsqrtf(var[col] + 1e-5f);
        float bb = beta[col] - mean[col] * g;
#pragma unroll
        for (int r = 0; r < 8; ++r)
            H[(r + hl * 8) * 64 + col] = fmaxf(c[r] * g + bb, 0.f);
    }
    __syncthreads();

    if (lane < 16) {
        float acc2 = b2[0];
        for (int j = 0; j < 64; ++j) acc2 += H[lane * 64 + j] * w2[j];
        logits[v0 + lane] = acc2;
    }
}

// ---- host-side orchestration ----------------------------------------------

extern "C" void kernel_launch(void* const* d_in, const int* in_sizes, int n_in,
                              void* d_out, int out_size, void* d_ws, size_t ws_size,
                              hipStream_t stream) {
    const float* x    = (const float*)d_in[0];
    const int*   ei   = (const int*)d_in[1];
    const float* la   = (const float*)d_in[2];
    const float* tse  = (const float*)d_in[3];
    const int*   bp   = (const int*)d_in[4];
    const int*   tt   = (const int*)d_in[5];
    const float* cr   = (const float*)d_in[6];
    const float* tsp  = (const float*)d_in[7];
    const float* tg   = (const float*)d_in[8];
    const float* r7   = (const float*)d_in[9];
    const float* r30  = (const float*)d_in[10];
    const int*   uts  = (const int*)d_in[11];
    const float* txe  = (const float*)d_in[12];
    const float* bke  = (const float*)d_in[13];
    const float* ee_w1 = (const float*)d_in[14]; const float* ee_b1 = (const float*)d_in[15];
    const float* ee_w2 = (const float*)d_in[16]; const float* ee_b2 = (const float*)d_in[17];
    const float* m1w = (const float*)d_in[18]; const float* m1b = (const float*)d_in[19];
    const float* u1w = (const float*)d_in[20]; const float* u1b = (const float*)d_in[21];
    const float* m2w = (const float*)d_in[22]; const float* m2b = (const float*)d_in[23];
    const float* u2w = (const float*)d_in[24]; const float* u2b = (const float*)d_in[25];
    const float* ecw1 = (const float*)d_in[26]; const float* ecb1 = (const float*)d_in[27];
    const float* ecw2 = (const float*)d_in[28]; const float* ecb2 = (const float*)d_in[29];
    const float* ncw1 = (const float*)d_in[30]; const float* ncb1 = (const float*)d_in[31];
    const float* bng = (const float*)d_in[32]; const float* bnb = (const float*)d_in[33];
    const float* bnm = (const float*)d_in[34]; const float* bnv = (const float*)d_in[35];
    const float* ncw2 = (const float*)d_in[36]; const float* ncb2 = (const float*)d_in[37];
    float* out = (float*)d_out;               // [NN node logits][E_TOT edge logits]

    // workspace carve-up
    char* w = (char*)d_ws;
    _Float16* eattr = (_Float16*)w; w += (size_t)E_TOT * 64 * sizeof(_Float16);
    float* probs = (float*)w;       w += (size_t)E_TOT * sizeof(float);
    float* sacc  = (float*)w;       w += (size_t)NN * 128 * sizeof(float);
    float* deg   = (float*)w;       w += (size_t)NN * sizeof(float);
    float* h1    = (float*)w;       w += (size_t)NN * 128 * sizeof(float);
    float* emb   = (float*)w;       w += (size_t)NN * 64 * sizeof(float);
    float* acc   = (float*)w;       w += (size_t)NN * 11 * sizeof(float);
    float* eagg  = (float*)w;       w += (size_t)NN * 8 * sizeof(float);
    int*   nowts = (int*)w;         w += 256;
    // packed f16 weight fragments (frag = 512 halves = 1KB each)
    const size_t FR = 512;
    _Float16* p_ee1 = (_Float16*)w; w += 16 * FR * 2;   // 2 ks x 8 nt
    _Float16* p_ee2 = (_Float16*)w; w += 16 * FR * 2;   // 4 x 4
    _Float16* p_m1  = (_Float16*)w; w += 48 * FR * 2;   // 6 x 8
    _Float16* p_u1  = (_Float16*)w; w += 64 * FR * 2;   // 8 x 8
    _Float16* p_m2  = (_Float16*)w; w += 48 * FR * 2;   // 6 x 8
    _Float16* p_u2  = (_Float16*)w; w += 32 * FR * 2;   // 8 x 4
    _Float16* p_ec1 = (_Float16*)w; w += 24 * FR * 2;   // 6 x 4
    _Float16* p_nc1 = (_Float16*)w; w += 12 * FR * 2;   // 3 x 4

    const int eb = (E_TOT + 255) / 256, nb = (NN + 255) / 256;

    // pre-pack weights into WMMA B-fragment layout (tiny, L2-resident)
    pack_w_k<<<16, 32, 0, stream>>>(ee_w1, 128,  38, 8, p_ee1);
    pack_w_k<<<16, 32, 0, stream>>>(ee_w2,  64, 128, 4, p_ee2);
    pack_w_k<<<48, 32, 0, stream>>>(m1w,   128, 192, 8, p_m1);
    pack_w_k<<<64, 32, 0, stream>>>(u1w,   128, 256, 8, p_u1);
    pack_w_k<<<48, 32, 0, stream>>>(m2w,   128, 192, 8, p_m2);
    pack_w_k<<<32, 32, 0, stream>>>(u2w,    64, 256, 4, p_u2);
    pack_w_k<<<24, 32, 0, stream>>>(ecw1,   64, 192, 4, p_ec1);
    pack_w_k<<<12, 32, 0, stream>>>(ncw1,   64,  72, 4, p_nc1);

    // zero accumulators (fresh every call; d_ws is not re-poisoned between replays)
    fset_k<<<(NN * 128 + 255) / 256, 256, 0, stream>>>(sacc, NN * 128, 0.f);
    fset_k<<<nb, 256, 0, stream>>>(deg, NN, 0.f);
    fset_k<<<(NN * 10 + 255) / 256, 256, 0, stream>>>(acc, NN * 10, 0.f);
    fset_k<<<nb, 256, 0, stream>>>(acc + 10 * NN, NN, 9999.f);
    fset_k<<<1, 256, 0, stream>>>((float*)nowts, 1, 0.f); // int 0 bits

    // 1. edge encoder
    edge_encoder_k<<<E_TOT / 16, 32, 0, stream>>>(la, cr, tsp, tg, r7, r30, tse, tt,
                                                  bp, txe, bke, p_ee1, ee_b1, p_ee2,
                                                  ee_b2, eattr);
    // 2. degree
    deg_k<<<eb, 256, 0, stream>>>(ei, deg);
    // 3. SAGE layer 1
    sage_msg_k<<<E_TOT / 16, 32, 0, stream>>>(x, eattr, ei, p_m1, m1b, sacc);
    sage_upd_k<<<NN / 16, 32, 0, stream>>>(x, sacc, deg, p_u1, u1b, 128, h1);
    // 4. SAGE layer 2
    fset_k<<<(NN * 128 + 255) / 256, 256, 0, stream>>>(sacc, NN * 128, 0.f);
    sage_msg_k<<<E_TOT / 16, 32, 0, stream>>>(h1, eattr, ei, p_m2, m2b, sacc);
    sage_upd_k<<<NN / 16, 32, 0, stream>>>(h1, sacc, deg, p_u2, u2b, 64, emb);
    // 5. edge classifier -> edge logits + probs
    edge_cls_k<<<E_TOT / 16, 32, 0, stream>>>(emb, eattr, ei, p_ec1, ecb1, ecw2, ecb2,
                                              out + NN, probs);
    // 6. edge-signal aggregation
    tsmax_k<<<eb, 256, 0, stream>>>(uts, nowts);
    eagg_k<<<eb, 256, 0, stream>>>(probs, ei, uts, nowts, acc);
    agg_fin_k<<<nb, 256, 0, stream>>>(acc, eagg);
    // 7. node classifier -> node logits
    node_cls_k<<<NN / 16, 32, 0, stream>>>(emb, eagg, ncw1 ? p_nc1 : p_nc1, ncb1,
                                           bng, bnb, bnm, bnv, ncw2, ncb2, out);
}
